// VesselGNN_57767310131222
// MI455X (gfx1250) — matrix operand order repaired and checked by reading, not compiled, not run on hardware
//
#include <hip/hip_runtime.h>

typedef __attribute__((ext_vector_type(16))) _Float16 v16h;
typedef __attribute__((ext_vector_type(8)))  _Float16 v8h;
typedef __attribute__((ext_vector_type(8)))  float    v8f;

#define BN_EPS 1e-5f

// ---------------- CSR build: count / dinv / scan / fill ----------------
__global__ void k_zero_int(int* p, int n) {
    int i = blockIdx.x * blockDim.x + threadIdx.x;
    if (i < n) p[i] = 0;
}

__global__ void k_count(const int* __restrict__ dst, int* cnt, int E) {
    int i = blockIdx.x * blockDim.x + threadIdx.x;
    if (i < E) atomicAdd(&cnt[dst[i]], 1);
}

// dinv[i] = rsqrt(deg) with deg = in-edges + 1 (self loop)
__global__ void k_dinv(const int* __restrict__ cnt, float* dinv, int n) {
    int i = blockIdx.x * blockDim.x + threadIdx.x;
    if (i < n) dinv[i] = rsqrtf((float)cnt[i] + 1.0f);
}

// per-256-block exclusive scan (LDS Hillis-Steele); blockSums optional
__global__ void k_scan_block(const int* __restrict__ in, int* __restrict__ outExcl,
                             int* __restrict__ blockSums, int n) {
    __shared__ int sm[256];
    int tid = threadIdx.x;
    int i = blockIdx.x * 256 + tid;
    int v = (i < n) ? in[i] : 0;
    sm[tid] = v;
    __syncthreads();
    for (int off = 1; off < 256; off <<= 1) {
        int t = (tid >= off) ? sm[tid - off] : 0;
        __syncthreads();
        sm[tid] += t;
        __syncthreads();
    }
    if (i < n) outExcl[i] = sm[tid] - v;       // exclusive
    if (tid == 255 && blockSums) blockSums[blockIdx.x] = sm[255];
}

__global__ void k_scan_add(const int* __restrict__ localExcl, const int* __restrict__ blockOff,
                           int* __restrict__ indptr, int n, int total) {
    int i = blockIdx.x * blockDim.x + threadIdx.x;
    if (i < n) indptr[i] = localExcl[i] + blockOff[i >> 8];
    if (i == 0) indptr[n] = total;
}

__global__ void k_fill(const int* __restrict__ src, const int* __restrict__ dst,
                       const int* __restrict__ indptr, int* __restrict__ fillpos,
                       const float* __restrict__ dinv,
                       int* __restrict__ esrc, float* __restrict__ enorm, int E) {
    int e = blockIdx.x * blockDim.x + threadIdx.x;
    if (e >= E) return;
    int s = src[e], d = dst[e];
    int p = indptr[d] + atomicAdd(&fillpos[d], 1);
    esrc[p]  = s;
    enorm[p] = dinv[s] * dinv[d];
}

// ---------------- conversions ----------------
__global__ void k_cvt_x(const float* __restrict__ x, _Float16* __restrict__ out,
                        int n, int cols, int colsPad) {
    int idx = blockIdx.x * blockDim.x + threadIdx.x;
    if (idx >= n * colsPad) return;
    int r = idx / colsPad, c = idx % colsPad;
    out[idx] = (c < cols) ? (_Float16)x[r * cols + c] : (_Float16)0.0f;
}

// W [K, Nout] f32 -> Wt [Nout, Kpad] f16 (transposed, zero K padding)
__global__ void k_cvt_wT(const float* __restrict__ W, _Float16* __restrict__ Wt,
                         int K, int Kpad, int Nout) {
    int idx = blockIdx.x * blockDim.x + threadIdx.x;
    if (idx >= Nout * Kpad) return;
    int o = idx / Kpad, kp = idx % Kpad;
    Wt[idx] = (kp < K) ? (_Float16)W[kp * Nout + o] : (_Float16)0.0f;
}

// ---------------- WMMA GEMM: C[M,Nout] = A[M,K](f16) * Bt[Nout,K]^T (+bias) ----------------
template <typename OUT_T>
__global__ void k_gemm_wmma(const _Float16* __restrict__ A, const _Float16* __restrict__ Bt,
                            const float* __restrict__ bias, OUT_T* __restrict__ C,
                            int M, int K, int Nout) {
    int wave = (int)((blockIdx.x * blockDim.x + threadIdx.x) >> 5);
    int lane = threadIdx.x & 31;
    int tilesN = Nout >> 4;
    int tileM = wave / tilesN;
    int tileN = wave % tilesN;
    if (tileM * 16 >= M) return;

    int hi  = lane >> 4;
    int l15 = lane & 15;
    int row = tileM * 16 + l15;
    int col = tileN * 16 + l15;

    const _Float16* ap = A  + (size_t)row * K + hi * 16;
    const _Float16* bp = Bt + (size_t)col * K + hi * 16;

    v8f acc = {};
    for (int kb = 0; kb < K; kb += 32) {
        v8h alo = *(const v8h*)(ap + kb);
        v8h ahi = *(const v8h*)(ap + kb + 8);
        v16h a = __builtin_shufflevector(alo, ahi, 0,1,2,3,4,5,6,7,8,9,10,11,12,13,14,15);
        v16h b = *(const v16h*)(bp + kb);
        acc = __builtin_amdgcn_wmma_f32_16x16x32_f16(false, a, false, b,
                                                     (short)0, acc, false, false);
    }

    int rbase = tileM * 16 + hi * 8;
    float bv = bias ? bias[col] : 0.0f;
#pragma unroll
    for (int r = 0; r < 8; ++r)
        C[(size_t)(rbase + r) * Nout + col] = (OUT_T)(acc[r] + bv);
}

// ---------------- pull-based aggregation + bias + BN + ReLU -> f16 ----------------
// one wave32 per destination node; VE = F/32 features per lane, f32 accumulation
template <int VE>
__global__ void k_aggregate(const _Float16* __restrict__ xwh, const int* __restrict__ indptr,
                            const int* __restrict__ esrc, const float* __restrict__ enorm,
                            const float* __restrict__ dinv,
                            const float* __restrict__ bb, const float* __restrict__ gg,
                            const float* __restrict__ bbe, const float* __restrict__ mm,
                            const float* __restrict__ vv,
                            _Float16* __restrict__ out, int n) {
    const int F = VE * 32;
    int node = (int)((blockIdx.x * blockDim.x + threadIdx.x) >> 5);
    int lane = threadIdx.x & 31;
    if (node >= n) return;

    int beg = indptr[node], end = indptr[node + 1];

    float acc[VE];
    float d2 = dinv[node] * dinv[node];                 // self-loop norm
    const _Float16* rp = xwh + (size_t)node * F + lane * VE;
#pragma unroll
    for (int k = 0; k < VE; ++k) acc[k] = d2 * (float)rp[k];

    for (int e = beg; e < end; ++e) {
        int s    = esrc[e];                              // wave-coherent
        float nr = enorm[e];
        const _Float16* sp = xwh + (size_t)s * F + lane * VE;
#pragma unroll
        for (int k = 0; k < VE; ++k) acc[k] += nr * (float)sp[k];
    }

#pragma unroll
    for (int k = 0; k < VE; ++k) {
        int j = lane * VE + k;
        float v = (acc[k] + bb[j] - mm[j]) * rsqrtf(vv[j] + BN_EPS) * gg[j] + bbe[j];
        out[(size_t)node * F + j] = (_Float16)fmaxf(v, 0.0f);
    }
}

// ---------------- segment pooling over sorted batch ----------------
__global__ void k_bounds_init(int* gstart, int* gend, int G, int n) {
    int i = blockIdx.x * blockDim.x + threadIdx.x;
    if (i < G) { gstart[i] = n; gend[i] = 0; }
}

__global__ void k_bounds(const int* __restrict__ batch, int* gstart, int* gend, int n) {
    int i = blockIdx.x * blockDim.x + threadIdx.x;
    if (i >= n) return;
    int g = batch[i];
    atomicMin(&gstart[g], i);
    atomicMax(&gend[g], i + 1);
}

// one block per graph; thread j owns feature j; streams the contiguous node range
__global__ void k_pool_seg(const _Float16* __restrict__ h, const int* __restrict__ gstart,
                           const int* __restrict__ gend, _Float16* __restrict__ pooled, int F) {
    int g = blockIdx.x;
    int j = threadIdx.x;
    int beg = gstart[g], end = gend[g];
    float sum = 0.0f, mx = 0.0f;                         // h >= 0 post-ReLU
    for (int i = beg; i < end; ++i) {
        float v = (float)h[(size_t)i * F + j];
        sum += v;
        mx = fmaxf(mx, v);
    }
    int c = end - beg; if (c < 0) c = 0;
    pooled[(size_t)g * 2 * F + j]     = (_Float16)(sum / fmaxf((float)c, 1.0f));
    pooled[(size_t)g * 2 * F + F + j] = (_Float16)mx;
}

// ---------------- host launch ----------------
extern "C" void kernel_launch(void* const* d_in, const int* in_sizes, int n_in,
                              void* d_out, int out_size, void* d_ws, size_t ws_size,
                              hipStream_t stream) {
    const float* x   = (const float*)d_in[0];
    const int*   ei  = (const int*)d_in[1];
    const int*   bat = (const int*)d_in[2];
    const float* W1  = (const float*)d_in[3];  const float* b1  = (const float*)d_in[4];
    const float* W2  = (const float*)d_in[5];  const float* b2  = (const float*)d_in[6];
    const float* W3  = (const float*)d_in[7];  const float* b3  = (const float*)d_in[8];
    const float* g1  = (const float*)d_in[9];  const float* be1 = (const float*)d_in[10];
    const float* rm1 = (const float*)d_in[11]; const float* rv1 = (const float*)d_in[12];
    const float* g2  = (const float*)d_in[13]; const float* be2 = (const float*)d_in[14];
    const float* rm2 = (const float*)d_in[15]; const float* rv2 = (const float*)d_in[16];
    const float* g3  = (const float*)d_in[17]; const float* be3 = (const float*)d_in[18];
    const float* rm3 = (const float*)d_in[19]; const float* rv3 = (const float*)d_in[20];
    const float* fcW = (const float*)d_in[21]; const float* fcb = (const float*)d_in[22];

    const int N = in_sizes[0] / 6;
    const int E = in_sizes[1] / 2;
    const int G = 32;
    const int* srcA = ei;
    const int* dstA = ei + E;
    const int NB = (N + 255) / 256;                      // scan blocks (157 <= 256)

    char* ws = (char*)d_ws;
    size_t off = 0;
    auto alloc = [&](size_t bytes) -> void* {
        void* p = ws + off;
        off = (off + bytes + 255) & ~(size_t)255;
        return p;
    };
    float*    dinv    = (float*)alloc((size_t)N * 4);
    int*      cnt     = (int*)alloc((size_t)N * 4);
    int*      fillpos = (int*)alloc((size_t)N * 4);
    int*      scanTmp = (int*)alloc((size_t)N * 4);
    int*      blkSum  = (int*)alloc(256 * 4);
    int*      blkOff  = (int*)alloc(256 * 4);
    int*      indptr  = (int*)alloc((size_t)(N + 1) * 4);
    int*      esrc    = (int*)alloc((size_t)E * 4);
    float*    enorm   = (float*)alloc((size_t)E * 4);
    _Float16* bufA    = (_Float16*)alloc((size_t)N * 128 * 2);
    _Float16* bufXWh  = (_Float16*)alloc((size_t)N * 128 * 2);
    _Float16* W1t     = (_Float16*)alloc(64 * 32 * 2);
    _Float16* W2t     = (_Float16*)alloc(128 * 64 * 2);
    _Float16* W3t     = (_Float16*)alloc(128 * 128 * 2);
    _Float16* fcWt    = (_Float16*)alloc(128 * 256 * 2);
    int*      gstart  = (int*)alloc(G * 4);
    int*      gend    = (int*)alloc(G * 4);
    _Float16* pooled  = (_Float16*)alloc(G * 256 * 2);

    const int T = 256;
    auto grid = [](long long tot, int bs) { return (int)((tot + bs - 1) / bs); };

    // ---- CSR build (once per call; ~1.3M int atomics + LDS scans, negligible) ----
    k_zero_int<<<grid(N, T), T, 0, stream>>>(cnt, N);
    k_zero_int<<<grid(N, T), T, 0, stream>>>(fillpos, N);
    k_count   <<<grid(E, T), T, 0, stream>>>(dstA, cnt, E);
    k_dinv    <<<grid(N, T), T, 0, stream>>>(cnt, dinv, N);
    k_scan_block<<<NB, 256, 0, stream>>>(cnt, scanTmp, blkSum, N);
    k_scan_block<<<1, 256, 0, stream>>>(blkSum, blkOff, nullptr, NB);
    k_scan_add<<<grid(N, T), T, 0, stream>>>(scanTmp, blkOff, indptr, N, E);
    k_fill    <<<grid(E, T), T, 0, stream>>>(srcA, dstA, indptr, fillpos, dinv, esrc, enorm, E);

    // ---- weight / input conversion ----
    k_cvt_x <<<grid((long long)N * 32, T), T, 0, stream>>>(x, bufA, N, 6, 32);
    k_cvt_wT<<<grid(64 * 32, T),   T, 0, stream>>>(W1,  W1t,  6,   32,  64);
    k_cvt_wT<<<grid(128 * 64, T),  T, 0, stream>>>(W2,  W2t,  64,  64,  128);
    k_cvt_wT<<<grid(128 * 128, T), T, 0, stream>>>(W3,  W3t,  128, 128, 128);
    k_cvt_wT<<<grid(128 * 256, T), T, 0, stream>>>(fcW, fcWt, 256, 256, 128);

    // ---- layers: WMMA GEMM (f16 out) + pull aggregation fused with bias/BN/ReLU ----
    int aggBlocks = grid((long long)N * 32, T);          // one wave per node, 8 waves/block

    // layer 1: K=32 (padded from 6), F=64
    {
        int tiles = (N / 16) * (64 / 16);
        k_gemm_wmma<_Float16><<<(tiles + 3) / 4, 128, 0, stream>>>(bufA, W1t, nullptr, bufXWh, N, 32, 64);
        k_aggregate<2><<<aggBlocks, T, 0, stream>>>(bufXWh, indptr, esrc, enorm, dinv,
                                                    b1, g1, be1, rm1, rv1, bufA, N);
    }
    // layer 2: K=64, F=128
    {
        int tiles = (N / 16) * (128 / 16);
        k_gemm_wmma<_Float16><<<(tiles + 3) / 4, 128, 0, stream>>>(bufA, W2t, nullptr, bufXWh, N, 64, 128);
        k_aggregate<4><<<aggBlocks, T, 0, stream>>>(bufXWh, indptr, esrc, enorm, dinv,
                                                    b2, g2, be2, rm2, rv2, bufA, N);
    }
    // layer 3: K=128, F=128
    {
        int tiles = (N / 16) * (128 / 16);
        k_gemm_wmma<_Float16><<<(tiles + 3) / 4, 128, 0, stream>>>(bufA, W3t, nullptr, bufXWh, N, 128, 128);
        k_aggregate<4><<<aggBlocks, T, 0, stream>>>(bufXWh, indptr, esrc, enorm, dinv,
                                                    b3, g3, be3, rm3, rv3, bufA, N);
    }

    // ---- pooling over sorted batch segments (no atomics in the feature pass) ----
    k_bounds_init<<<1, G, 0, stream>>>(gstart, gend, G, N);
    k_bounds     <<<grid(N, T), T, 0, stream>>>(bat, gstart, gend, N);
    k_pool_seg   <<<G, 128, 0, stream>>>(bufA, gstart, gend, pooled, 128);

    // ---- final FC via WMMA: [32,256] x [256,128] + fcb -> d_out (f32) ----
    int tilesFC = (G / 16) * (128 / 16);
    k_gemm_wmma<float><<<(tilesFC + 3) / 4, 128, 0, stream>>>(pooled, fcWt, fcb, (float*)d_out, G, 256, 128);
}
